// Pointnet2MSG_43654047596730
// MI455X (gfx1250) — compile-verified, matrix operand order checked by reading, not run on hardware
//
#include <hip/hip_runtime.h>
#include <hip/hip_bf16.h>

typedef __attribute__((ext_vector_type(16))) _Float16 v16h;
typedef __attribute__((ext_vector_type(8)))  _Float16 v8h;
typedef __attribute__((ext_vector_type(8)))  float    v8f;

// ---------------------------------------------------------------------------
// Farthest point sampling: one block per batch. dist[] lives in global ws.
// Stable argmax (first occurrence) to match jnp.argmax semantics.
// ---------------------------------------------------------------------------
__global__ void fps_kernel(const float* __restrict__ xyz, int Np, int npoint,
                           int* __restrict__ fidx, float* __restrict__ distws) {
  const int NT = 256;
  int b = blockIdx.x, tid = threadIdx.x;
  const float* X = xyz + (size_t)b * Np * 3;
  float* dist = distws + (size_t)b * Np;
  __shared__ float sval[NT];
  __shared__ int   sidx[NT];
  for (int i = tid; i < Np; i += NT) dist[i] = 1e10f;
  if (tid == 0) fidx[b * npoint] = 0;
  int last = 0;
  __syncthreads();
  for (int s = 1; s < npoint; ++s) {
    float px = X[last * 3 + 0], py = X[last * 3 + 1], pz = X[last * 3 + 2];
    float best = -1.0f; int bi = 0x7fffffff;
    for (int i = tid; i < Np; i += NT) {
      float dx = X[i * 3 + 0] - px;
      float dy = X[i * 3 + 1] - py;
      float dz = X[i * 3 + 2] - pz;
      float d = dx * dx + dy * dy + dz * dz;
      float dm = dist[i];
      dm = d < dm ? d : dm;
      dist[i] = dm;
      if (dm > best) { best = dm; bi = i; }
    }
    sval[tid] = best; sidx[tid] = bi;
    __syncthreads();
    for (int stp = NT / 2; stp > 0; stp >>= 1) {
      if (tid < stp) {
        float v2 = sval[tid + stp]; int i2 = sidx[tid + stp];
        if (v2 > sval[tid] || (v2 == sval[tid] && i2 < sidx[tid])) {
          sval[tid] = v2; sidx[tid] = i2;
        }
      }
      __syncthreads();
    }
    last = sidx[0];
    if (tid == 0) fidx[b * npoint + s] = last;
    __syncthreads();
  }
}

// dst[b,s,c] = src[b, idx[b,s], c]
__global__ void gather_points_kernel(const float* __restrict__ src, const int* __restrict__ idx,
                                     float* __restrict__ dst, int Np, int S, int C, int NB) {
  int t = blockIdx.x * blockDim.x + threadIdx.x;
  int total = NB * S * C;
  if (t >= total) return;
  int c = t % C, s = (t / C) % S, b = t / (C * S);
  dst[t] = src[((size_t)b * Np + idx[b * S + s]) * C + c];
}

// Ball query: smallest `ns` indices with d2 < r2, padded with first hit.
__global__ void ball_query_kernel(const float* __restrict__ xyz, const float* __restrict__ new_xyz,
                                  int Np, int S, int ns, float r2, int NB,
                                  int* __restrict__ out) {
  int t = blockIdx.x * blockDim.x + threadIdx.x;
  if (t >= NB * S) return;
  int b = t / S;
  const float* c = new_xyz + (size_t)t * 3;
  const float* X = xyz + (size_t)b * Np * 3;
  int buf[32];
  int cnt = 0;
  for (int j = 0; j < Np && cnt < ns; ++j) {
    float dx = X[j * 3 + 0] - c[0];
    float dy = X[j * 3 + 1] - c[1];
    float dz = X[j * 3 + 2] - c[2];
    float d2 = dx * dx + dy * dy + dz * dz;
    if (d2 < r2) buf[cnt++] = j;
  }
  int first = cnt > 0 ? buf[0] : 0;
  for (int k = cnt; k < ns; ++k) buf[k] = first;
  int* o = out + (size_t)t * ns;
  for (int k = 0; k < ns; ++k) o[k] = buf[k];
}

// out[b,s,k,:] = [xyz[idx]-ctr, feats[idx]]
__global__ void group_gather_kernel(const float* __restrict__ xyz, const float* __restrict__ feats,
                                    const float* __restrict__ new_xyz, const int* __restrict__ idx,
                                    int Np, int S, int ns, int C, int NB,
                                    float* __restrict__ out) {
  int t = blockIdx.x * blockDim.x + threadIdx.x;
  int total = NB * S * ns;
  if (t >= total) return;
  int k = t % ns; int s = (t / ns) % S; int b = t / (ns * S);
  int j = idx[((size_t)b * S + s) * ns + k];
  const float* src = xyz + ((size_t)b * Np + j) * 3;
  const float* ctr = new_xyz + ((size_t)b * S + s) * 3;
  float* o = out + (size_t)t * (3 + C);
  o[0] = src[0] - ctr[0];
  o[1] = src[1] - ctr[1];
  o[2] = src[2] - ctr[2];
  if (feats) {
    const float* f = feats + ((size_t)b * Np + j) * C;
    for (int c = 0; c < C; ++c) o[3 + c] = f[c];
  }
}

// ---------------------------------------------------------------------------
// Fused linear layer via WMMA: Out = act((A @ W) * g + b)
// A: [M,K] f32 row-major (M must be a multiple of 128), W: [K,N] f32.
// Block = 256 threads = 8 waves; 128x64 output tile. Each wave owns 16 rows x
// 64 cols = one A fragment reused across four v_wmma_f32_16x16x32_f16.
//
// Pipelining: LDS is double-buffered; global loads for tile kt+1 are staged
// into registers BEFORE the ds_load fragments + WMMAs of tile kt, and flushed
// to the other LDS buffer afterwards. One __syncthreads per k-step.
//
// Fragment trick: the WMMA 16-bit fragment k-permutation
//   pos(k) = (k&7) | ((k&16)>>1) | ((k&8)<<1)
// is applied when STORING tiles to LDS, so each lane's 16 fragment halves are
// contiguous and load as two ds_load_b128.
//   A frag (lane L): m = L&15, halves = As[m][ (L>=16)*16 .. +15 ]
//   B frag (lane L): n = L&15, halves = Bs[n][ (L>=16)*16 .. +15 ]
// ---------------------------------------------------------------------------
__global__ void wmma_linear_kernel(const float* __restrict__ A, const float* __restrict__ W,
                                   const float* __restrict__ g, const float* __restrict__ bias,
                                   float* __restrict__ Out, int M, int K, int N, int relu) {
  __shared__ _Float16 As[2][128 * 40];  // [buf][row][posk], stride 40 halves
  __shared__ _Float16 Bs[2][64 * 40];   // [buf][col][posk], stride 40 halves
  int tid = threadIdx.x, lane = tid & 31, wid = tid >> 5;
  int m0 = blockIdx.x * 128, n0 = blockIdx.y * 64;
  v8f acc0 = {}, acc1 = {}, acc2 = {}, acc3 = {};

  // staging coordinates: fixed column per thread, row strided -> coalesced
  const int cA  = tid & 31, rA0 = tid >> 5;              // A: k-col, row base
  const int posA = (cA & 7) | ((cA & 16) >> 1) | ((cA & 8) << 1);
  const int nB  = tid & 63, kB0 = tid >> 6;              // W: n-col, k base
  const bool nin = (n0 + nB) < N;
  int kt_max = (K + 31) / 32;

  float va[16];   // A-tile register stage (16 rows per thread)
  float vw[8];    // W-tile register stage (8 k per thread)

  auto stage_regs = [&](int kt) {
    int kbase = kt * 32;
    if (kbase + 32 <= K) {             // uniform fast path
      const float* Ap = A + (size_t)(m0 + rA0) * K + kbase + cA;
#pragma unroll
      for (int i = 0; i < 16; ++i) va[i] = Ap[(size_t)i * 8 * K];
    } else {
      bool kin = (kbase + cA) < K;
#pragma unroll
      for (int i = 0; i < 16; ++i)
        va[i] = kin ? A[(size_t)(m0 + rA0 + i * 8) * K + kbase + cA] : 0.0f;
    }
    if (kbase + 32 <= K && nin) {
      const float* Wp = W + (size_t)(kbase + kB0) * N + n0 + nB;
#pragma unroll
      for (int i = 0; i < 8; ++i) vw[i] = Wp[(size_t)i * 4 * N];
    } else {
#pragma unroll
      for (int i = 0; i < 8; ++i) {
        int k = kbase + kB0 + i * 4;
        vw[i] = (k < K && nin) ? W[(size_t)k * N + n0 + nB] : 0.0f;
      }
    }
  };
  auto store_lds = [&](int buf) {
#pragma unroll
    for (int i = 0; i < 16; ++i)
      As[buf][(rA0 + i * 8) * 40 + posA] = (_Float16)va[i];
#pragma unroll
    for (int i = 0; i < 8; ++i) {
      int k = kB0 + i * 4;
      int posk = (k & 7) | ((k & 16) >> 1) | ((k & 8) << 1);
      Bs[buf][nB * 40 + posk] = (_Float16)vw[i];
    }
  };

  stage_regs(0);
  store_lds(0);
  for (int kt = 0; kt < kt_max; ++kt) {
    __syncthreads();                   // buf (kt&1) fully written
    int cur = kt & 1;
    if (kt + 1 < kt_max) stage_regs(kt + 1);   // overlap HBM with WMMA below

    int koff = (lane >> 4) * 16;       // contiguous 16 halves per lane
    const v8h* ap = (const v8h*)&As[cur][(wid * 16 + (lane & 15)) * 40 + koff];
    v16h af = __builtin_shufflevector(ap[0], ap[1], 0, 1, 2, 3, 4, 5, 6, 7,
                                      8, 9, 10, 11, 12, 13, 14, 15);
    const v8h* bp0 = (const v8h*)&Bs[cur][((lane & 15) + 0)  * 40 + koff];
    const v8h* bp1 = (const v8h*)&Bs[cur][((lane & 15) + 16) * 40 + koff];
    const v8h* bp2 = (const v8h*)&Bs[cur][((lane & 15) + 32) * 40 + koff];
    const v8h* bp3 = (const v8h*)&Bs[cur][((lane & 15) + 48) * 40 + koff];
    v16h bf0 = __builtin_shufflevector(bp0[0], bp0[1], 0, 1, 2, 3, 4, 5, 6, 7,
                                       8, 9, 10, 11, 12, 13, 14, 15);
    v16h bf1 = __builtin_shufflevector(bp1[0], bp1[1], 0, 1, 2, 3, 4, 5, 6, 7,
                                       8, 9, 10, 11, 12, 13, 14, 15);
    v16h bf2 = __builtin_shufflevector(bp2[0], bp2[1], 0, 1, 2, 3, 4, 5, 6, 7,
                                       8, 9, 10, 11, 12, 13, 14, 15);
    v16h bf3 = __builtin_shufflevector(bp3[0], bp3[1], 0, 1, 2, 3, 4, 5, 6, 7,
                                       8, 9, 10, 11, 12, 13, 14, 15);
    acc0 = __builtin_amdgcn_wmma_f32_16x16x32_f16(false, af, false, bf0,
                                                  (short)0, acc0, false, false);
    acc1 = __builtin_amdgcn_wmma_f32_16x16x32_f16(false, af, false, bf1,
                                                  (short)0, acc1, false, false);
    acc2 = __builtin_amdgcn_wmma_f32_16x16x32_f16(false, af, false, bf2,
                                                  (short)0, acc2, false, false);
    acc3 = __builtin_amdgcn_wmma_f32_16x16x32_f16(false, af, false, bf3,
                                                  (short)0, acc3, false, false);

    if (kt + 1 < kt_max) store_lds((kt + 1) & 1);   // other buffer: no race
  }

  // C/D layout: VGPR v, lane L -> m = v + 8*(L>=16), n = L&15
  int mbase = m0 + wid * 16 + (lane >> 4) * 8;   // M % 128 == 0 at all call sites
  int nbase = n0 + (lane & 15);
#pragma unroll
  for (int t = 0; t < 4; ++t) {
    int n = nbase + 16 * t;
    if (n < N) {
      float gs = g ? g[n] : 1.0f;
      float bb = bias ? bias[n] : 0.0f;
      v8f acc = (t == 0) ? acc0 : (t == 1) ? acc1 : (t == 2) ? acc2 : acc3;
#pragma unroll
      for (int v = 0; v < 8; ++v) {
        float y = acc[v] * gs + bb;
        if (relu) y = y > 0.0f ? y : 0.0f;
        Out[(size_t)(mbase + v) * N + n] = y;
      }
    }
  }
}

// out[row, coff+c] = max_k h[row*ns+k, c]   (row = b*S+s, out stride Ctot)
__global__ void maxpool_concat_kernel(const float* __restrict__ h, float* __restrict__ out,
                                      int S, int ns, int C, int Ctot, int coff, int NB) {
  int t = blockIdx.x * blockDim.x + threadIdx.x;
  int total = NB * S * C;
  if (t >= total) return;
  int c = t % C; int row = t / C;
  const float* hp = h + ((size_t)row * ns) * C + c;
  float m = hp[0];
  for (int k = 1; k < ns; ++k) {
    float v = hp[(size_t)k * C];
    m = v > m ? v : m;
  }
  out[(size_t)row * Ctot + coff + c] = m;
}

// Three nearest neighbors + normalized inverse-distance weights.
__global__ void three_nn_kernel(const float* __restrict__ unknown, const float* __restrict__ known,
                                int Nun, int Nk, int NB,
                                int* __restrict__ idx3, float* __restrict__ w3) {
  int t = blockIdx.x * blockDim.x + threadIdx.x;
  if (t >= NB * Nun) return;
  int b = t / Nun;
  const float* u = unknown + (size_t)t * 3;
  const float* Kp = known + (size_t)b * Nk * 3;
  float d0 = 1e30f, d1 = 1e30f, d2v = 1e30f;
  int i0 = 0, i1 = 0, i2 = 0;
  for (int j = 0; j < Nk; ++j) {
    float dx = Kp[j * 3 + 0] - u[0];
    float dy = Kp[j * 3 + 1] - u[1];
    float dz = Kp[j * 3 + 2] - u[2];
    float d = dx * dx + dy * dy + dz * dz;
    if (d < d0)      { d2v = d1; i2 = i1; d1 = d0; i1 = i0; d0 = d; i0 = j; }
    else if (d < d1) { d2v = d1; i2 = i1; d1 = d; i1 = j; }
    else if (d < d2v){ d2v = d; i2 = j; }
  }
  float w0 = 1.0f / (d0 + 1e-8f), w1 = 1.0f / (d1 + 1e-8f), w2 = 1.0f / (d2v + 1e-8f);
  float s = w0 + w1 + w2;
  idx3[t * 3 + 0] = i0; idx3[t * 3 + 1] = i1; idx3[t * 3 + 2] = i2;
  w3[t * 3 + 0] = w0 / s; w3[t * 3 + 1] = w1 / s; w3[t * 3 + 2] = w2 / s;
}

// X[row, c] = sum_i w_i * feats[b, idx_i, c]   (X stride ldx)
__global__ void interp_apply_kernel(const float* __restrict__ feats, const int* __restrict__ idx3,
                                    const float* __restrict__ w3, float* __restrict__ X,
                                    int Nun, int Nk, int C, int ldx, int NB) {
  int t = blockIdx.x * blockDim.x + threadIdx.x;
  int total = NB * Nun * C;
  if (t >= total) return;
  int c = t % C; int row = t / C; int b = row / Nun;
  const float* f = feats + (size_t)b * Nk * C;
  int i0 = idx3[row * 3 + 0], i1 = idx3[row * 3 + 1], i2 = idx3[row * 3 + 2];
  float v = w3[row * 3 + 0] * f[(size_t)i0 * C + c]
          + w3[row * 3 + 1] * f[(size_t)i1 * C + c]
          + w3[row * 3 + 2] * f[(size_t)i2 * C + c];
  X[(size_t)row * ldx + c] = v;
}

// dst[r, coff+c] = src[r, c]  (dst stride ldx) — skip-feature concat
__global__ void copy_chan_kernel(const float* __restrict__ src, float* __restrict__ dst,
                                 int rows, int C, int ldx, int coff) {
  int t = blockIdx.x * blockDim.x + threadIdx.x;
  if (t >= rows * C) return;
  int c = t % C, r = t / C;
  dst[(size_t)r * ldx + coff + c] = src[(size_t)r * C + c];
}

// ---------------------------------------------------------------------------
// Host orchestration
// ---------------------------------------------------------------------------
struct LayerP { const float *W, *g, *b; int cin, cout; };

extern "C" void kernel_launch(void* const* d_in, const int* in_sizes, int n_in,
                              void* d_out, int out_size, void* d_ws, size_t ws_size,
                              hipStream_t stream) {
  (void)in_sizes; (void)n_in; (void)out_size; (void)ws_size;
  const int NB = 4, NPTS = 8192;
  const int LEV[3] = {2048, 512, 128};
  const float R2[3][2] = {{0.01f * 0.01f, 0.03f * 0.03f},
                          {0.025f * 0.025f, 0.05f * 0.05f},
                          {0.1f * 0.1f, 0.15f * 0.15f}};
  const int NS[3][2] = {{16, 32}, {16, 32}, {16, 32}};
  static const int SA_D[3][2][4] = {{{3, 16, 16, 32}, {3, 32, 32, 64}},
                                    {{99, 64, 64, 128}, {99, 64, 96, 128}},
                                    {{259, 128, 256, 512}, {259, 128, 256, 512}}};
  static const int FP_D[3][3] = {{128, 128, 128}, {352, 256, 128}, {1280, 256, 256}};

  // --- input mapping (dict insertion order: pointcloud, sa, fp, fc_W, fc_b;
  //     each mlp layer contributes W, g, b in that order) ---
  const float* pc = (const float*)d_in[0];
  int p = 1;
  LayerP sa[3][2][3];
  for (int i = 0; i < 3; ++i)
    for (int j = 0; j < 2; ++j)
      for (int l = 0; l < 3; ++l) {
        sa[i][j][l].cin = SA_D[i][j][l];
        sa[i][j][l].cout = SA_D[i][j][l + 1];
        sa[i][j][l].W = (const float*)d_in[p++];
        sa[i][j][l].g = (const float*)d_in[p++];
        sa[i][j][l].b = (const float*)d_in[p++];
      }
  LayerP fp[3][2];
  for (int i = 0; i < 3; ++i)
    for (int l = 0; l < 2; ++l) {
      fp[i][l].cin = FP_D[i][l];
      fp[i][l].cout = FP_D[i][l + 1];
      fp[i][l].W = (const float*)d_in[p++];
      fp[i][l].g = (const float*)d_in[p++];
      fp[i][l].b = (const float*)d_in[p++];
    }
  const float* fcW = (const float*)d_in[p++];
  const float* fcb = (const float*)d_in[p++];

  // --- workspace bump allocator (needs ~200MB) ---
  char* ws = (char*)d_ws;
  size_t off = 0;
  auto allocf = [&](size_t n) -> float* {
    float* r = (float*)(ws + off);
    off = (off + n * sizeof(float) + 255) & ~(size_t)255;
    return r;
  };
  auto alloci = [&](size_t n) -> int* {
    int* r = (int*)(ws + off);
    off = (off + n * sizeof(int) + 255) & ~(size_t)255;
    return r;
  };
  float* distws  = allocf((size_t)NB * NPTS);
  int*   fidx    = alloci((size_t)NB * 2048);
  int*   qidx    = alloci((size_t)NB * 2048 * 32);
  float* xyz1    = allocf((size_t)NB * 2048 * 3);
  float* xyz2    = allocf((size_t)NB * 512 * 3);
  float* xyz3    = allocf((size_t)NB * 128 * 3);
  float* feats1  = allocf((size_t)NB * 2048 * 96);
  float* feats2  = allocf((size_t)NB * 512 * 256);
  float* feats3  = allocf((size_t)NB * 128 * 1024);
  float* feats2b = allocf((size_t)NB * 512 * 256);
  float* feats1b = allocf((size_t)NB * 2048 * 128);
  float* feats0  = allocf((size_t)NB * 8192 * 128);
  int*   i3idx   = alloci((size_t)NB * 8192 * 3);
  float* i3w     = allocf((size_t)NB * 8192 * 3);
  float* Gbuf    = allocf((size_t)6488064);     // max grouped tensor (L1 j1)
  float* pingA   = allocf((size_t)16777216);    // max 262144 x 64
  float* pingB   = allocf((size_t)16777216);

  auto cdiv = [](int a, int b) { return (a + b - 1) / b; };
  auto gemm = [&](const float* Ab, const LayerP& L, float* Ob, int Mrows, int relu) {
    dim3 gr(cdiv(Mrows, 128), cdiv(L.cout, 64));
    wmma_linear_kernel<<<gr, 256, 0, stream>>>(Ab, L.W, L.g, L.b, Ob,
                                               Mrows, L.cin, L.cout, relu);
  };

  const float* l_xyz[4]  = {pc, xyz1, xyz2, xyz3};
  float*       l_feat[4] = {nullptr, feats1, feats2, feats3};
  const int NpL[4] = {8192, 2048, 512, 128};
  const int CfL[4] = {0, 96, 256, 1024};

  // ---------------- Set abstraction levels ----------------
  for (int i = 0; i < 3; ++i) {
    int S = LEV[i], Np = NpL[i], Cin = CfL[i], Ctot = CfL[i + 1];
    fps_kernel<<<NB, 256, 0, stream>>>(l_xyz[i], Np, S, fidx, distws);
    gather_points_kernel<<<cdiv(NB * S * 3, 256), 256, 0, stream>>>(
        l_xyz[i], fidx, (float*)l_xyz[i + 1], Np, S, 3, NB);
    int coff = 0;
    for (int j = 0; j < 2; ++j) {
      int ns = NS[i][j];
      ball_query_kernel<<<cdiv(NB * S, 256), 256, 0, stream>>>(
          l_xyz[i], l_xyz[i + 1], Np, S, ns, R2[i][j], NB, qidx);
      group_gather_kernel<<<cdiv(NB * S * ns, 256), 256, 0, stream>>>(
          l_xyz[i], l_feat[i], l_xyz[i + 1], qidx, Np, S, ns, Cin, NB, Gbuf);
      int Mrows = NB * S * ns;
      gemm(Gbuf,  sa[i][j][0], pingA, Mrows, 1);
      gemm(pingA, sa[i][j][1], pingB, Mrows, 1);
      gemm(pingB, sa[i][j][2], pingA, Mrows, 1);
      int C3 = sa[i][j][2].cout;
      maxpool_concat_kernel<<<cdiv(NB * S * C3, 256), 256, 0, stream>>>(
          pingA, l_feat[i + 1], S, ns, C3, Ctot, coff, NB);
      coff += C3;
    }
  }

  // ---------------- Feature propagation ----------------
  // i = 3: 128 -> 512 points, concat skip(feats2, 256), mlp fp[2]
  {
    int Nun = 512, Nk = 128, C = 1024, Cs = 256, Kt = 1280;
    three_nn_kernel<<<cdiv(NB * Nun, 256), 256, 0, stream>>>(xyz2, xyz3, Nun, Nk, NB, i3idx, i3w);
    interp_apply_kernel<<<cdiv(NB * Nun * C, 256), 256, 0, stream>>>(
        feats3, i3idx, i3w, pingA, Nun, Nk, C, Kt, NB);
    copy_chan_kernel<<<cdiv(NB * Nun * Cs, 256), 256, 0, stream>>>(
        feats2, pingA, NB * Nun, Cs, Kt, C);
    gemm(pingA, fp[2][0], pingB, NB * Nun, 1);
    gemm(pingB, fp[2][1], feats2b, NB * Nun, 1);
  }
  // i = 2: 512 -> 2048 points, concat skip(feats1, 96), mlp fp[1]
  {
    int Nun = 2048, Nk = 512, C = 256, Cs = 96, Kt = 352;
    three_nn_kernel<<<cdiv(NB * Nun, 256), 256, 0, stream>>>(xyz1, xyz2, Nun, Nk, NB, i3idx, i3w);
    interp_apply_kernel<<<cdiv(NB * Nun * C, 256), 256, 0, stream>>>(
        feats2b, i3idx, i3w, pingA, Nun, Nk, C, Kt, NB);
    copy_chan_kernel<<<cdiv(NB * Nun * Cs, 256), 256, 0, stream>>>(
        feats1, pingA, NB * Nun, Cs, Kt, C);
    gemm(pingA, fp[1][0], pingB, NB * Nun, 1);
    gemm(pingB, fp[1][1], feats1b, NB * Nun, 1);
  }
  // i = 1: 2048 -> 8192 points, no skip, mlp fp[0]
  {
    int Nun = 8192, Nk = 2048, C = 128, Kt = 128;
    three_nn_kernel<<<cdiv(NB * Nun, 256), 256, 0, stream>>>(pc, xyz1, Nun, Nk, NB, i3idx, i3w);
    interp_apply_kernel<<<cdiv(NB * Nun * C, 256), 256, 0, stream>>>(
        feats1b, i3idx, i3w, pingA, Nun, Nk, C, Kt, NB);
    gemm(pingA, fp[0][0], pingB, NB * Nun, 1);
    gemm(pingB, fp[0][1], feats0, NB * Nun, 1);
  }

  // ---------------- Final FC: feats0 @ fc_W + fc_b ----------------
  {
    LayerP fc{fcW, nullptr, fcb, 128, 8};
    gemm(feats0, fc, (float*)d_out, NB * NPTS, 0);
  }
}
